// GATNet_35407710388607
// MI455X (gfx1250) — matrix operand order repaired and checked by reading, not compiled, not run on hardware
//
#include <hip/hip_runtime.h>
#include <hip/hip_bf16.h>

// ---------------------------------------------------------------------------
// 2-layer GAT for gfx1250 (MI455X).
// Dense GEMMs -> v_wmma_f32_16x16x32_f16, A tile staged in LDS via
// global_load_async_to_lds_b128 (+ s_wait_asynccnt), ds_load_b128 fragment
// reads.  Edge softmax/aggregation -> coalesced float4 atomics.
// ---------------------------------------------------------------------------

typedef __attribute__((ext_vector_type(16))) _Float16 v16h;
typedef __attribute__((ext_vector_type(8)))  float    v8f;
typedef int v4i __attribute__((vector_size(16)));   // matches async-LDS builtin

#define AS1 __attribute__((address_space(1)))
#define AS3 __attribute__((address_space(3)))

#if defined(__has_builtin)
#if __has_builtin(__builtin_amdgcn_global_load_async_to_lds_b128)
#define ASYNC_COPY 1
#endif
#if __has_builtin(__builtin_amdgcn_s_wait_asynccnt)
#define HAVE_WAIT_ASYNC 1
#endif
#endif
#ifndef ASYNC_COPY
#define ASYNC_COPY 0
#endif
#ifndef HAVE_WAIT_ASYNC
#define HAVE_WAIT_ASYNC 0
#endif

static constexpr int NN   = 50000;   // nodes
static constexpr int NE   = 800000;  // edges (before self loops)
static constexpr int F0   = 256;     // in_dim
static constexpr int H1H  = 8;       // heads layer 1
static constexpr int H1C  = 32;      // channels/head layer 1
static constexpr int F1   = 256;     // H1H*H1C
static constexpr int F2   = 16;      // out_dim

// ---------------- helpers ----------------
__device__ __forceinline__ unsigned f2s(float f) {
    unsigned u = __float_as_uint(f);
    return (u & 0x80000000u) ? ~u : (u | 0x80000000u);
}
__device__ __forceinline__ float s2f(unsigned u) {
    return (u & 0x80000000u) ? __uint_as_float(u & 0x7FFFFFFFu)
                             : __uint_as_float(~u);
}
__device__ __forceinline__ float lrelu(float v) { return v > 0.0f ? v : 0.2f * v; }

__device__ __forceinline__ void wait_async_copies() {
#if ASYNC_COPY
#if HAVE_WAIT_ASYNC
    __builtin_amdgcn_s_wait_asynccnt(0);
#else
    asm volatile("s_wait_asynccnt 0x0" ::: "memory");
#endif
#endif
}

// ---------------- conversion / layout prep ----------------
__global__ void k_f32_to_f16(const float* __restrict__ in, _Float16* __restrict__ out, int n) {
    int i = blockIdx.x * blockDim.x + threadIdx.x;
    if (i < n) out[i] = (_Float16)in[i];
}

// W[K][N] (row-major) -> Wt[N][K] f16
__global__ void k_transpose_to_f16(const float* __restrict__ W, _Float16* __restrict__ Wt,
                                   int K, int N) {
    int i = blockIdx.x * blockDim.x + threadIdx.x;
    if (i >= K * N) return;
    int n = i / K, k = i % K;
    Wt[i] = (_Float16)W[(size_t)k * N + n];
}

// ---------------- WMMA GEMM with LDS-staged A tile ----------------
// C[MxN] = A[MxK] * Bt[NxK]^T.  One block per 16-row M tile; the block stages
// the 16xK f16 A tile into LDS once (async-to-LDS), then each wave loops over
// 16-col N tiles, reading A fragments via ds_load_b128 and streaming B
// fragments from global (W matrices are L2-resident on MI455X).
template <int KDIM>
__launch_bounds__(256)
__global__ void k_gemm_wmma(const _Float16* __restrict__ A,
                            const _Float16* __restrict__ Bt,
                            float* __restrict__ C,
                            int N, int tiles_n) {
    constexpr int LDSROW = KDIM + 8;                 // padded row stride (halves)
    __shared__ alignas(16) _Float16 Atile[16 * LDSROW];

    const int tm     = blockIdx.x;
    const int tid    = threadIdx.x;
    const int nwaves = blockDim.x >> 5;
    const int wave   = tid >> 5;
    const int lane   = tid & 31;
    const int l      = lane & 15, half = lane >> 4;

    // ---- stage A tile (16 rows x KDIM halves) in 16-byte chunks ----
    constexpr int CHUNKS_PER_ROW = (KDIM * 2) / 16;  // 32 for K=256
    constexpr int NCHUNK = 16 * CHUNKS_PER_ROW;      // 512
    for (int c = tid; c < NCHUNK; c += blockDim.x) {
        int row  = c / CHUNKS_PER_ROW;
        int col8 = (c % CHUNKS_PER_ROW) * 8;         // in halves
        const _Float16* gsrc = A + (size_t)(tm * 16 + row) * KDIM + col8;
        _Float16*       ldst = &Atile[row * LDSROW + col8];
#if ASYNC_COPY
        __builtin_amdgcn_global_load_async_to_lds_b128(
            (AS1 v4i*)gsrc, (AS3 v4i*)ldst, 0, 0);
#else
        *(uint4*)ldst = *(const uint4*)gsrc;
#endif
    }
    wait_async_copies();
    __syncthreads();

    for (int tn = wave; tn < tiles_n; tn += nwaves) {
        const _Float16* brow = Bt + (size_t)(tn * 16 + l) * KDIM;
        v8f acc = {};
        for (int kk = 0; kk < KDIM; kk += 32) {
            __builtin_prefetch(brow + kk + 64, 0, 1);      // global_prefetch_b8
            union { unsigned u[8]; v16h v; } a, b;
            // A 16x32 fragment: lane = M row; K 0..7/8..15 per lane half in
            // VGPR 0..3, K 16..23/24..31 in VGPR 4..7 -> two ds_load_b128.
            const _Float16* ar = &Atile[l * LDSROW + kk + 8 * half];
            *(uint4*)&a.u[0] = *(const uint4*)ar;
            *(uint4*)&a.u[4] = *(const uint4*)(ar + 16);
            // B 32x16 fragment: lane = N col; K 0..7/8..15 per lane half in
            // dwords 0..3, K 16..23/24..31 in dwords 4..7 (matches A).
#pragma unroll
            for (int j = 0; j < 8; ++j)
                b.u[j] = *(const unsigned*)(brow + kk + 8 * half + 2 * j + (j >= 4 ? 8 : 0));

            acc = __builtin_amdgcn_wmma_f32_16x16x32_f16(
                false, a.v, false, b.v, (short)0, acc, false, false);
        }
        // C/D layout: VGPR v -> M = v + 8*half, N = l
#pragma unroll
        for (int v = 0; v < 8; ++v)
            C[(size_t)(tm * 16 + v + 8 * half) * N + tn * 16 + l] = acc[v];
    }
}

// ---------------- attention logits: al = sum_c h[n,h,c]*a[h,c] ----------------
__global__ void k_attn_logits(const float* __restrict__ Hm,
                              const float* __restrict__ a_s, const float* __restrict__ a_d,
                              float* __restrict__ al_s, float* __restrict__ al_d,
                              int n, int Hh, int Cc) {
    int idx = blockIdx.x * blockDim.x + threadIdx.x;
    if (idx >= n * Hh) return;
    int h = idx % Hh;
    const float* hp = Hm + (size_t)idx * Cc;   // (node*Hh + h)*Cc
    float ss = 0.f, sd = 0.f;
    for (int c = 0; c < Cc; ++c) {
        float v = hp[c];
        ss += v * a_s[h * Cc + c];
        sd += v * a_d[h * Cc + c];
    }
    al_s[idx] = ss;
    al_d[idx] = sd;
}

// ---------------- edge pass 1: segment max ----------------
__global__ void k_edge_max(const long long* __restrict__ ei,
                           const float* __restrict__ al_s, const float* __restrict__ al_d,
                           unsigned* __restrict__ mU, int Hh) {
    int idx = blockIdx.x * blockDim.x + threadIdx.x;
    int total = (NE + NN) * Hh;
    if (idx >= total) return;
    int e = idx / Hh, h = idx % Hh;
    long long s, d;
    if (e < NE) { s = ei[e]; d = ei[NE + e]; } else { s = d = e - NE; }
    float v = lrelu(al_s[s * Hh + h] + al_d[d * Hh + h]);
    atomicMax(&mU[d * Hh + h], f2s(v));
}

// ---------------- edge pass 2: denom = segment sum of exp(e - m) ----------------
__global__ void k_edge_denom(const long long* __restrict__ ei,
                             const float* __restrict__ al_s, const float* __restrict__ al_d,
                             const unsigned* __restrict__ mU, float* __restrict__ denom,
                             int Hh) {
    int idx = blockIdx.x * blockDim.x + threadIdx.x;
    int total = (NE + NN) * Hh;
    if (idx >= total) return;
    int e = idx / Hh, h = idx % Hh;
    long long s, d;
    if (e < NE) { s = ei[e]; d = ei[NE + e]; } else { s = d = e - NE; }
    float v = lrelu(al_s[s * Hh + h] + al_d[d * Hh + h]);
    atomicAdd(&denom[d * Hh + h], __expf(v - s2f(mU[d * Hh + h])));
}

// ---------------- edge pass 3: out[dst] += alpha * h[src] (coalesced) --------
// One thread per (edge, 4 contiguous channels): adjacent lanes touch adjacent
// floats -> full-line global_load_b128 reads + dense global_atomic_add_f32.
__global__ void k_edge_agg(const long long* __restrict__ ei,
                           const float* __restrict__ al_s, const float* __restrict__ al_d,
                           const unsigned* __restrict__ mU, const float* __restrict__ denom,
                           const float* __restrict__ Hm, float* __restrict__ out,
                           int Hh, int Cc) {
    const int F = Hh * Cc, G = F >> 2;               // 4-channel groups per edge
    int idx = blockIdx.x * blockDim.x + threadIdx.x;
    int total = (NE + NN) * G;
    if (idx >= total) return;
    int e = idx / G, g = idx % G;
    long long s, d;
    if (e < NE) { s = ei[e]; d = ei[NE + e]; } else { s = d = e - NE; }
    int h = (g * 4) / Cc;
    float v = lrelu(al_s[s * Hh + h] + al_d[d * Hh + h]);
    float alpha = __expf(v - s2f(mU[d * Hh + h])) / denom[d * Hh + h];
    float4 hv = *(const float4*)(Hm + (size_t)s * F + g * 4);
    float* op = out + (size_t)d * F + g * 4;
    atomicAdd(op + 0, alpha * hv.x);
    atomicAdd(op + 1, alpha * hv.y);
    atomicAdd(op + 2, alpha * hv.z);
    atomicAdd(op + 3, alpha * hv.w);
}

// ---------------- bias + relu -> f16 (feeds GEMM2) ----------------
__global__ void k_bias_relu_f16(const float* __restrict__ agg, const float* __restrict__ b,
                                _Float16* __restrict__ outh, int n, int F) {
    int i = blockIdx.x * blockDim.x + threadIdx.x;
    if (i >= n * F) return;
    float v = agg[i] + b[i % F];
    outh[i] = (_Float16)(v > 0.f ? v : 0.f);
}

// ---------------- bias + log_softmax (in place, F <= 16) ----------------
__global__ void k_bias_logsoftmax(float* __restrict__ io, const float* __restrict__ b,
                                  int n, int F) {
    int node = blockIdx.x * blockDim.x + threadIdx.x;
    if (node >= n) return;
    float v[16];
    float mx = -3.0e38f;
    for (int j = 0; j < F; ++j) {
        v[j] = io[(size_t)node * F + j] + b[j];
        mx = fmaxf(mx, v[j]);
    }
    float s = 0.f;
    for (int j = 0; j < F; ++j) s += __expf(v[j] - mx);
    float ls = __logf(s) + mx;
    for (int j = 0; j < F; ++j) io[(size_t)node * F + j] = v[j] - ls;
}

// ---------------------------------------------------------------------------
extern "C" void kernel_launch(void* const* d_in, const int* in_sizes, int n_in,
                              void* d_out, int out_size, void* d_ws, size_t ws_size,
                              hipStream_t stream) {
    const float*     x    = (const float*)d_in[0];
    const long long* ei   = (const long long*)d_in[1];   // [2, NE] int64
    const float*     W1   = (const float*)d_in[2];
    const float*     a_s1 = (const float*)d_in[3];
    const float*     a_d1 = (const float*)d_in[4];
    const float*     b1   = (const float*)d_in[5];
    const float*     W2   = (const float*)d_in[6];
    const float*     a_s2 = (const float*)d_in[7];
    const float*     a_d2 = (const float*)d_in[8];
    const float*     b2   = (const float*)d_in[9];
    float* out = (float*)d_out;

    char* ws = (char*)d_ws;
    size_t off = 0;
    auto take = [&](size_t bytes) -> char* {
        char* p = ws + off;
        off += (bytes + 255) & ~(size_t)255;
        return p;
    };
    _Float16* xh     = (_Float16*)take((size_t)NN * F0 * 2);   // reused as h2 f16
    _Float16* w1t    = (_Float16*)take((size_t)F1 * F0 * 2);
    _Float16* w2t    = (_Float16*)take((size_t)F2 * F1 * 2);
    float*    H1b    = (float*)take((size_t)NN * F1 * 4);
    float*    agg1   = (float*)take((size_t)NN * F1 * 4);
    float*    al_s1b = (float*)take((size_t)NN * H1H * 4);
    float*    al_d1b = (float*)take((size_t)NN * H1H * 4);
    unsigned* m1u    = (unsigned*)take((size_t)NN * H1H * 4);
    float*    den1   = (float*)take((size_t)NN * H1H * 4);
    float*    H2b    = (float*)take((size_t)NN * F2 * 4);
    float*    al_s2b = (float*)take((size_t)NN * 4);
    float*    al_d2b = (float*)take((size_t)NN * 4);
    unsigned* m2u    = (unsigned*)take((size_t)NN * 4);
    float*    den2   = (float*)take((size_t)NN * 4);
    _Float16* h2h    = xh;   // reuse: x f16 not needed after GEMM1

    const int TPB = 256;
    auto blks = [](long long n) { return (unsigned)((n + 255) / 256); };

    // ---- prep: f16 conversions / weight transposes ----
    k_f32_to_f16<<<blks((long long)NN * F0), TPB, 0, stream>>>(x, xh, NN * F0);
    k_transpose_to_f16<<<blks(F0 * F1), TPB, 0, stream>>>(W1, w1t, F0, F1);
    k_transpose_to_f16<<<blks(F1 * F2), TPB, 0, stream>>>(W2, w2t, F1, F2);

    // ---- layer 1: GEMM (WMMA, 8 waves/block over 16 N tiles) ----
    k_gemm_wmma<F0><<<NN / 16, 256, 0, stream>>>(xh, w1t, H1b, F1, F1 / 16);

    k_attn_logits<<<blks((long long)NN * H1H), TPB, 0, stream>>>(
        H1b, a_s1, a_d1, al_s1b, al_d1b, NN, H1H, H1C);

    (void)hipMemsetAsync(m1u,  0, (size_t)NN * H1H * 4, stream);
    (void)hipMemsetAsync(den1, 0, (size_t)NN * H1H * 4, stream);
    (void)hipMemsetAsync(agg1, 0, (size_t)NN * F1 * 4, stream);

    long long etot1 = (long long)(NE + NN) * H1H;
    k_edge_max  <<<blks(etot1), TPB, 0, stream>>>(ei, al_s1b, al_d1b, m1u, H1H);
    k_edge_denom<<<blks(etot1), TPB, 0, stream>>>(ei, al_s1b, al_d1b, m1u, den1, H1H);
    long long atot1 = (long long)(NE + NN) * (F1 / 4);
    k_edge_agg  <<<blks(atot1), TPB, 0, stream>>>(ei, al_s1b, al_d1b, m1u, den1,
                                                  H1b, agg1, H1H, H1C);

    k_bias_relu_f16<<<blks((long long)NN * F1), TPB, 0, stream>>>(agg1, b1, h2h, NN, F1);

    // ---- layer 2: GEMM (WMMA, 1 wave/block, single N tile) ----
    k_gemm_wmma<F1><<<NN / 16, 32, 0, stream>>>(h2h, w2t, H2b, F2, F2 / 16);

    k_attn_logits<<<blks(NN), TPB, 0, stream>>>(H2b, a_s2, a_d2, al_s2b, al_d2b, NN, 1, F2);

    (void)hipMemsetAsync(m2u,  0, (size_t)NN * 4, stream);
    (void)hipMemsetAsync(den2, 0, (size_t)NN * 4, stream);
    (void)hipMemsetAsync(out,  0, (size_t)NN * F2 * 4, stream);

    long long etot2 = (long long)(NE + NN);
    k_edge_max  <<<blks(etot2), TPB, 0, stream>>>(ei, al_s2b, al_d2b, m2u, 1);
    k_edge_denom<<<blks(etot2), TPB, 0, stream>>>(ei, al_s2b, al_d2b, m2u, den2, 1);
    long long atot2 = (long long)(NE + NN) * (F2 / 4);
    k_edge_agg  <<<blks(atot2), TPB, 0, stream>>>(ei, al_s2b, al_d2b, m2u, den2,
                                                  H2b, out, 1, F2);

    k_bias_logsoftmax<<<blks(NN), TPB, 0, stream>>>(out, b2, NN, F2);
}
